// FSNet_51135880626536
// MI455X (gfx1250) — matrix operand order repaired and checked by reading, not compiled, not run on hardware
//
#include <hip/hip_runtime.h>
#include <hip/hip_bf16.h>
#include <math.h>
#include <stdint.h>

// ---------------------------------------------------------------------------
// CDNA5 / gfx1250 WMMA types
// ---------------------------------------------------------------------------
typedef __attribute__((ext_vector_type(16))) __bf16 v16bf;
typedef __attribute__((ext_vector_type(8)))  float  v8f;

union FragB16 { v16bf v; unsigned u[8]; };

#define HID   256
#define G3    768        // 3*HID
#define BATCH 128
#define BSTRIDE 40       // padded bf16 row stride for LDS weight stages (80B: 16B-aligned, conflict-free)
#define HBS    264       // padded bf16 row stride for LDS h (528B = 132 dwords -> lanes spread over banks)

__device__ __forceinline__ float sigmoidf_(float x) { return 1.0f / (1.0f + __expf(-x)); }
__device__ __forceinline__ float seluf_(float x) {
  const float sc = 1.0507009873554805f, al = 1.6732632423543772f;
  return x > 0.f ? sc * x : sc * al * (__expf(x) - 1.f);
}

// gfx1250 async global->LDS copy (ASYNCcnt-tracked), 16B per lane.
__device__ __forceinline__ void async_copy16(void* lds_dst, const void* gsrc) {
  unsigned lds_off = (unsigned)(uintptr_t)lds_dst;
  unsigned long long ga = (unsigned long long)(uintptr_t)gsrc;
  asm volatile("global_load_async_to_lds_b128 %0, %1, off"
               :: "v"(lds_off), "v"(ga)
               : "memory");
}
__device__ __forceinline__ void wait_async0() {
  asm volatile("s_wait_asynccnt 0" ::: "memory");
}

// ---------------------------------------------------------------------------
// Embedding gather: seq_bf16[b*T+t][d] = bf16(emb[flow[b*T+t]][d])
// ---------------------------------------------------------------------------
__global__ void k_gather(const int* __restrict__ flow, const float* __restrict__ emb,
                         __bf16* __restrict__ out, int rows, int dim) {
  int idx = blockIdx.x * blockDim.x + threadIdx.x;
  if (idx >= rows * dim) return;
  int r = idx / dim, d = idx - r * dim;
  out[idx] = (__bf16)emb[(size_t)flow[r] * dim + d];
}

// ---------------------------------------------------------------------------
// Weight prep: f32 (K x N, row-major) -> bf16 transposed (N x K, row-major)
// ---------------------------------------------------------------------------
__global__ void k_transpose_bf16(const float* __restrict__ src, __bf16* __restrict__ dst,
                                 int K, int N) {
  int idx = blockIdx.x * blockDim.x + threadIdx.x;
  if (idx >= K * N) return;
  int k = idx / N, n = idx - k * N;
  dst[(size_t)n * K + k] = (__bf16)src[idx];
}

// ---------------------------------------------------------------------------
// WMMA bf16 GEMM, double-buffered async-LDS weight staging + pipelined A.
//   C[M x N] (f32) = act( A(bf16, row-major, lda) x Bt(bf16, N rows of K) + bias )
// Block = 256 threads (8 waves) covering 128 rows x 64 cols.
// ---------------------------------------------------------------------------
__global__ __launch_bounds__(256) void k_gemm_bf16(
    const __bf16* __restrict__ A, int lda, int rowdiv,
    const __bf16* __restrict__ Bt, int K, int N,
    const float* __restrict__ bias,
    float* __restrict__ C, int ldc,
    __bf16* __restrict__ Cb, int ldcb, int act) {
  __shared__ __bf16 bstage[2][64][BSTRIDE];

  const int wave = threadIdx.x >> 5;
  const int lane = threadIdx.x & 31;
  const int lm   = lane & 15;
  const int grp  = lane >> 4;
  const int rowBase = blockIdx.y * 128 + wave * 16;
  const int colBase = blockIdx.x * 64;

  // staging assignment: thread -> (row n in tile, 8-element k segment)
  const int sn = threadIdx.x >> 2;        // 0..63
  const int sk = (threadIdx.x & 3) * 8;   // 0,8,16,24
  const int brow = colBase + sn;
  const int browc = brow < N ? brow : (N - 1);   // clamp: cols >= N never stored
  const __bf16* bsrc = Bt + (size_t)browc * K + sk;

  v8f acc[4];
#pragma unroll
  for (int t = 0; t < 4; ++t)
#pragma unroll
    for (int e = 0; e < 8; ++e) acc[t][e] = 0.f;

  const int arow = (rowBase + lm) / rowdiv;
  const __bf16* aptr = A + (size_t)arow * lda;

  // prologue: stage first weight chunk, load first A fragment
  async_copy16(&bstage[0][sn][sk], bsrc);
  FragB16 a;
#pragma unroll
  for (int v = 0; v < 8; ++v) {
    int kb = 2 * v + grp * 16;
    a.u[v] = *(const unsigned*)(aptr + kb);
  }
  wait_async0();
  __syncthreads();

  int cur = 0;
  for (int kc = 0; kc < K; kc += 32) {
    FragB16 an = a;
    if (kc + 32 < K) {
      // prefetch next weight chunk (async->LDS) and next A fragment (regs)
      async_copy16(&bstage[cur ^ 1][sn][sk], bsrc + kc + 32);
#pragma unroll
      for (int v = 0; v < 8; ++v) {
        int kb = 2 * v + grp * 16 + kc + 32;
        an.u[v] = *(const unsigned*)(aptr + kb);
      }
    }
#pragma unroll
    for (int t = 0; t < 4; ++t) {
      FragB16 b;
      const __bf16* bp = &bstage[cur][t * 16 + lm][grp * 16];
#pragma unroll
      for (int v = 0; v < 8; ++v) b.u[v] = *(const unsigned*)(bp + 2 * v);
      acc[t] = __builtin_amdgcn_wmma_f32_16x16x32_bf16(false, a.v, false, b.v,
                                                       (short)0, acc[t], false, false);
    }
    wait_async0();     // own async writes done
    __syncthreads();   // all waves' writes done + all reads of cur done
    a = an;
    cur ^= 1;
  }

#pragma unroll
  for (int t = 0; t < 4; ++t) {
    int c = colBase + t * 16 + lm;
    if (c >= N) continue;
    float bv = bias ? bias[c] : 0.f;
#pragma unroll
    for (int e = 0; e < 8; ++e) {
      int r = rowBase + e + grp * 8;
      float vv = acc[t][e] + bv;
      if (act == 1) vv = seluf_(vv);
      C[(size_t)r * ldc + c] = vv;
      if (Cb) Cb[(size_t)r * ldcb + c] = (__bf16)vv;
    }
  }
}

// ---------------------------------------------------------------------------
// Persistent bidirectional GRU scan. grid.x = 2 (0 = fw, 1 = bw),
// block = 1024 threads (32 waves). LDS: h f32 (update) + padded h bf16
// (WMMA A operand, conflict-free stride) + double-buffered U stage filled by
// global_load_async_to_lds_b128 (48KB chunk pulled once per WGP per K-chunk,
// cyclically prefetched since U is step-invariant).
// Each wave owns (m-tile, 4 j-tiles) and computes the z/r/h~ column triple
// per j-tile so gate math stays in accumulator layout.
// ---------------------------------------------------------------------------
__global__ __launch_bounds__(1024) void k_gru_scan(
    const float* __restrict__ xp_fw, const float* __restrict__ xp_bw,
    const __bf16* __restrict__ Ut_fw, const __bf16* __restrict__ Ut_bw,
    const float* __restrict__ br_fw, const float* __restrict__ br_bw,
    float* __restrict__ ys, __bf16* __restrict__ ysb, int ld_ys,
    __bf16* __restrict__ hT, int ld_hT, int T) {
  extern __shared__ char smem[];
  float*  hf  = (float*)smem;                                    // 128*256 f32
  __bf16* hb  = (__bf16*)(smem + BATCH * HID * sizeof(float));   // 128*HBS bf16 (padded)
  __bf16* ust = (__bf16*)(smem + BATCH * HID * sizeof(float)
                               + BATCH * HBS * sizeof(__bf16));  // 2*768*BSTRIDE bf16

  const int dir = blockIdx.x;
  const float*  xp = dir ? xp_bw : xp_fw;
  const __bf16* Ut = dir ? Ut_bw : Ut_fw;
  const float*  br = dir ? br_bw : br_fw;
  ys += dir * HID;
  if (ysb) ysb += dir * HID;
  hT += dir * HID;

  const int tid = threadIdx.x;
  for (int i = tid; i < BATCH * HID; i += 1024) hf[i] = 0.f;
  for (int i = tid; i < BATCH * HBS; i += 1024) hb[i] = (__bf16)0.f;

  // U staging map: 768 rows x 4 segments of 16B; 3 segments per thread
  int  sn_[3], sk_[3];
#pragma unroll
  for (int p = 0; p < 3; ++p) {
    int s = tid + p * 1024;
    sn_[p] = s >> 2;
    sk_[p] = (s & 3) * 8;
  }
  // prologue: stage chunk 0 into buffer 0
#pragma unroll
  for (int p = 0; p < 3; ++p)
    async_copy16(&ust[(0 * G3 + sn_[p]) * BSTRIDE + sk_[p]],
                 Ut + (size_t)sn_[p] * HID + sk_[p]);
  wait_async0();
  __syncthreads();

  const int wave = tid >> 5, lane = tid & 31, lm = lane & 15, grp = lane >> 4;
  const int mtile = wave & 7;   // 8 row tiles  (128 rows)
  const int ng    = wave >> 3;  // 4 groups of 4 j-tiles (256 cols)
  const int arow  = mtile * 16 + lm;

  // hoist recurrent biases (step-invariant)
  float bz[4], brr[4], bh[4];
#pragma unroll
  for (int tr = 0; tr < 4; ++tr) {
    int j = (ng * 4 + tr) * 16 + lm;
    bz[tr] = br[j]; brr[tr] = br[256 + j]; bh[tr] = br[512 + j];
  }

  int cur = 0;
  float hn[4][8];
  for (int s = 0; s < T; ++s) {
    const int tt = dir ? (T - 1 - s) : s;

    v8f acc[4][3];
#pragma unroll
    for (int t = 0; t < 4; ++t)
#pragma unroll
      for (int gte = 0; gte < 3; ++gte)
#pragma unroll
        for (int e = 0; e < 8; ++e) acc[t][gte][e] = 0.f;

#pragma unroll
    for (int kci = 0; kci < 8; ++kci) {
      const int kc = kci * 32;
      // cyclic prefetch of the next U chunk (chunk 7 refills chunk 0 for next step)
      const int nk = ((kci + 1) & 7) * 32;
#pragma unroll
      for (int p = 0; p < 3; ++p)
        async_copy16(&ust[((cur ^ 1) * G3 + sn_[p]) * BSTRIDE + sk_[p]],
                     Ut + (size_t)sn_[p] * HID + nk + sk_[p]);

      FragB16 a;
      const __bf16* hrow = hb + arow * HBS;
#pragma unroll
      for (int v = 0; v < 8; ++v) {
        int kb = 2 * v + grp * 16 + kc;
        a.u[v] = *(const unsigned*)(hrow + kb);   // conflict-free ds_load
      }
#pragma unroll
      for (int tr = 0; tr < 4; ++tr) {
        int jt = ng * 4 + tr;
#pragma unroll
        for (int gte = 0; gte < 3; ++gte) {       // z / r / h~ column triple
          int n = (gte * 16 + jt) * 16 + lm;      // row of Ut chunk in [0,768)
          FragB16 b;
          const __bf16* bp = &ust[(cur * G3 + n) * BSTRIDE + grp * 16];
#pragma unroll
          for (int v = 0; v < 8; ++v) b.u[v] = *(const unsigned*)(bp + 2 * v);
          acc[tr][gte] = __builtin_amdgcn_wmma_f32_16x16x32_bf16(
              false, a.v, false, b.v, (short)0, acc[tr][gte], false, false);
        }
      }
      wait_async0();
      __syncthreads();
      cur ^= 1;
    }

    // Gate math: accumulator element (tr, e) maps to (batch row r, hidden col j).
#pragma unroll
    for (int tr = 0; tr < 4; ++tr) {
      int j = (ng * 4 + tr) * 16 + lm;
#pragma unroll
      for (int e = 0; e < 8; ++e) {
        int r = mtile * 16 + e + grp * 8;
        const float* xr_ = xp + ((size_t)r * T + tt) * G3;
        float z  = sigmoidf_(xr_[j]       + acc[tr][0][e] + bz[tr]);
        float rg = sigmoidf_(xr_[256 + j] + acc[tr][1][e] + brr[tr]);
        float hh = tanhf(xr_[512 + j] + rg * (acc[tr][2][e] + bh[tr]));
        float hold = hf[r * HID + j];
        hn[tr][e] = z * hold + (1.f - z) * hh;
      }
    }
    __syncthreads();   // everyone done reading old h
#pragma unroll
    for (int tr = 0; tr < 4; ++tr) {
      int j = (ng * 4 + tr) * 16 + lm;
#pragma unroll
      for (int e = 0; e < 8; ++e) {
        int r = mtile * 16 + e + grp * 8;
        float v = hn[tr][e];
        hf[r * HID + j] = v;
        hb[r * HBS + j] = (__bf16)v;
        ys[((size_t)r * T + tt) * ld_ys + j] = v;
        if (ysb) ysb[((size_t)r * T + tt) * ld_ys + j] = (__bf16)v;
      }
    }
    __syncthreads();   // h updated before next step's WMMA reads
  }

  // final hidden state -> feature buffer (bf16)
#pragma unroll
  for (int tr = 0; tr < 4; ++tr) {
    int j = (ng * 4 + tr) * 16 + lm;
#pragma unroll
    for (int e = 0; e < 8; ++e) {
      int r = mtile * 16 + e + grp * 8;
      hT[(size_t)r * ld_hT + j] = (__bf16)hn[tr][e];
    }
  }
}

// ---------------------------------------------------------------------------
// argmax over class logits; pred stored as float(index)
// ---------------------------------------------------------------------------
__global__ void k_argmax(const float* __restrict__ logits, float* __restrict__ pred,
                         int B, int C) {
  int b = blockIdx.x * blockDim.x + threadIdx.x;
  if (b >= B) return;
  int best = 0;
  float bv = logits[(size_t)b * C];
  for (int c = 1; c < C; ++c) {
    float v = logits[(size_t)b * C + c];
    if (v > bv) { bv = v; best = c; }
  }
  pred[b] = (float)best;
}

// ---------------------------------------------------------------------------
// Host orchestration
// ---------------------------------------------------------------------------
extern "C" void kernel_launch(void* const* d_in, const int* in_sizes, int n_in,
                              void* d_out, int out_size, void* d_ws, size_t ws_size,
                              hipStream_t stream) {
  (void)in_sizes; (void)n_in; (void)out_size; (void)ws_size;
  const int B = 128, T = 128, H = 256, LEN_DIM = 128, LEN_NUM = 2000, CLS = 50;
  const int BT = B * T;

  const int*   flow = (const int*)d_in[0];
  const float* emb  = (const float*)d_in[2];

  // GRU params: enc0, enc1, dec0, dec1; each {fw(W,U,b), bw(W,U,b)}
  struct GruP { const float *W, *U, *b; int din; };
  GruP g[8];
  const int din_[4] = {128, 512, 1024, 512};
  for (int L = 0; L < 4; ++L)
    for (int d = 0; d < 2; ++d) {
      int i = 3 + L * 6 + d * 3;
      g[L * 2 + d] = { (const float*)d_in[i], (const float*)d_in[i + 1],
                       (const float*)d_in[i + 2], din_[L] };
    }
  const float* recW1   = (const float*)d_in[27];
  const float* recb1   = (const float*)d_in[28];
  const float* recW2   = (const float*)d_in[29];
  const float* recb2   = (const float*)d_in[30];
  const float* compW   = (const float*)d_in[31];
  const float* compbia = (const float*)d_in[32];
  const float* clsW    = (const float*)d_in[33];
  const float* clsbia  = (const float*)d_in[34];

  float* out = (float*)d_out;

  // --- scratch carve ---
  size_t off = 0;
  auto alloc = [&](size_t bytes) -> void* {
    void* p = (char*)d_ws + off;
    off += (bytes + 255) & ~(size_t)255;
    return p;
  };
  __bf16* seqb = (__bf16*)alloc((size_t)BT * LEN_DIM * 2);
  __bf16* wt[8]; __bf16* ut[8];
  for (int j = 0; j < 8; ++j) wt[j] = (__bf16*)alloc((size_t)G3 * g[j].din * 2);
  for (int j = 0; j < 8; ++j) ut[j] = (__bf16*)alloc((size_t)G3 * H * 2);
  __bf16* rw1t = (__bf16*)alloc((size_t)H * 1024 * 2);
  __bf16* rw2t = (__bf16*)alloc((size_t)LEN_NUM * H * 2);
  __bf16* cwt  = (__bf16*)alloc((size_t)512 * 2048 * 2);
  __bf16* clt  = (__bf16*)alloc((size_t)CLS * 512 * 2);
  float*  xpf  = (float*)alloc((size_t)BT * G3 * 4);
  float*  xpb  = (float*)alloc((size_t)BT * G3 * 4);
  float*  enc_cur   = (float*)alloc((size_t)BT * 512 * 4);
  __bf16* enc_curb  = (__bf16*)alloc((size_t)BT * 512 * 2);
  float*  dec_outs  = (float*)alloc((size_t)BT * 1024 * 4);
  __bf16* dec_outsb = (__bf16*)alloc((size_t)BT * 1024 * 2);
  __bf16* featb = (__bf16*)alloc((size_t)B * 2048 * 2);   // [e_fea | d_fea]
  float*  rhid  = (float*)alloc((size_t)BT * H * 4);
  __bf16* rhidb = (__bf16*)alloc((size_t)BT * H * 2);
  float*  compf = (float*)alloc((size_t)B * 512 * 4);
  __bf16* compb_= (__bf16*)alloc((size_t)B * 512 * 2);

  auto tr = [&](const float* src, __bf16* dst, int K, int N) {
    int tot = K * N;
    k_transpose_bf16<<<(tot + 255) / 256, 256, 0, stream>>>(src, dst, K, N);
  };
  auto gemm = [&](const __bf16* A, int lda, int rowdiv, const __bf16* Bt,
                  int K, int N, int M, const float* bias,
                  float* C, int ldc, __bf16* Cb, int ldcb, int act) {
    dim3 grid((N + 63) / 64, M / 128);
    k_gemm_bf16<<<grid, 256, 0, stream>>>(A, lda, rowdiv, Bt, K, N, bias, C, ldc, Cb, ldcb, act);
  };
  // LDS: hf(128*256*4) + hb(128*HBS*2) + ustage(2*768*BSTRIDE*2) = 321536B <= 320KB
  const size_t smemScan = (size_t)BATCH * HID * 4 + (size_t)BATCH * HBS * 2
                        + (size_t)2 * G3 * BSTRIDE * 2;
  auto scan = [&](int j0, float* ysf, __bf16* ysb_, int ld, __bf16* hTp) {
    k_gru_scan<<<2, 1024, smemScan, stream>>>(
        xpf, xpb, ut[j0], ut[j0 + 1], g[j0].b + G3, g[j0 + 1].b + G3,
        ysf, ysb_, ld, hTp, 2048, T);
  };

  // 1) embedding gather (f32 -> bf16)
  k_gather<<<(BT * LEN_DIM + 255) / 256, 256, 0, stream>>>(flow, emb, seqb, BT, LEN_DIM);

  // 2) weight prep (transpose + bf16)
  for (int j = 0; j < 8; ++j) { tr(g[j].W, wt[j], g[j].din, G3); tr(g[j].U, ut[j], H, G3); }
  tr(recW1, rw1t, 1024, H);
  tr(recW2, rw2t, H, LEN_NUM);
  tr(compW, cwt, 2048, 512);
  tr(clsW,  clt, 512, CLS);

  // 3) encoder layer 0
  gemm(seqb, LEN_DIM, 1, wt[0], LEN_DIM, G3, BT, g[0].b, xpf, G3, nullptr, 0, 0);
  gemm(seqb, LEN_DIM, 1, wt[1], LEN_DIM, G3, BT, g[1].b, xpb, G3, nullptr, 0, 0);
  scan(0, enc_cur, enc_curb, 512, featb + 0);
  // encoder layer 1
  gemm(enc_curb, 512, 1, wt[2], 512, G3, BT, g[2].b, xpf, G3, nullptr, 0, 0);
  gemm(enc_curb, 512, 1, wt[3], 512, G3, BT, g[3].b, xpb, G3, nullptr, 0, 0);
  scan(2, enc_cur, enc_curb, 512, featb + 512);

  // 4) decoder layer 0: input = e_fea broadcast over T (rowdiv = T)
  gemm(featb, 2048, T, wt[4], 1024, G3, BT, g[4].b, xpf, G3, nullptr, 0, 0);
  gemm(featb, 2048, T, wt[5], 1024, G3, BT, g[5].b, xpb, G3, nullptr, 0, 0);
  scan(4, dec_outs, dec_outsb, 1024, featb + 1024);
  // decoder layer 1
  gemm(dec_outsb, 1024, 1, wt[6], 512, G3, BT, g[6].b, xpf, G3, nullptr, 0, 0);
  gemm(dec_outsb, 1024, 1, wt[7], 512, G3, BT, g[7].b, xpb, G3, nullptr, 0, 0);
  scan(6, dec_outs + 512, dec_outsb + 512, 1024, featb + 1536);

  // 5) reconstruction head: selu(dec_out @ W1 + b1) @ W2 + b2
  gemm(dec_outsb, 1024, 1, rw1t, 1024, H, BT, recb1, rhid, H, rhidb, H, 1);
  gemm(rhidb, H, 1, rw2t, H, LEN_NUM, BT, recb2, out + B * CLS + B, LEN_NUM, nullptr, 0, 0);

  // 6) classifier head: feature = selu([e_fea|d_fea] @ comp_W + b); logits = feature @ cls_W + b
  gemm(featb, 2048, 1, cwt, 2048, 512, B, compbia, compf, 512, compb_, 512, 1);
  gemm(compb_, 512, 1, clt, 512, CLS, B, clsbia, out, CLS, nullptr, 0, 0);

  // 7) pred = argmax(logits)
  k_argmax<<<1, 128, 0, stream>>>(out, out + B * CLS, B, CLS);
}
